// GATv2_33500744909178
// MI455X (gfx1250) — compile-verified, matrix operand order checked by reading, not run on hardware
//
#include <hip/hip_runtime.h>

#define N_NODES   50000
#define N_EDGES   1600000
#define EP        (N_EDGES + N_NODES)   // edges + self loops = 1,650,000 (divisible by 16)
#define F_NODE    128
#define F_EDGE    32
#define OUTF      128
#define N_LAYERS  3
#define N_CLASSES 10
#define N_GRAPHS  512
#define SLOPE     0.2f

typedef __attribute__((ext_vector_type(16))) __bf16 bf16x16;
typedef __attribute__((ext_vector_type(8)))  __bf16 bf16x8;
typedef __attribute__((ext_vector_type(8)))  float  f32x8;

// Build a v16bf A/B operand from the two 8-element chunks the 16-bit WMMA
// layout wants per lane (lanes 0-15: K 0-7 & 16-23; lanes 16-31: K 8-15 & 24-31).
__device__ __forceinline__ bf16x16 load_bf16x16_split(const __bf16* p, int base) {
  bf16x8 lo = *(const bf16x8*)(p + base);
  bf16x8 hi = *(const bf16x8*)(p + base + 16);
  bf16x16 r;
#pragma unroll
  for (int i = 0; i < 8; ++i) { r[i] = lo[i]; r[i + 8] = hi[i]; }
  return r;
}

// Order-preserving float <-> uint map for atomicMax-based segment max.
__device__ __forceinline__ unsigned mono(float f) {
  unsigned u = __float_as_uint(f);
  return (u & 0x80000000u) ? ~u : (u | 0x80000000u);
}
__device__ __forceinline__ float demono(unsigned u) {
  return __uint_as_float((u & 0x80000000u) ? (u & 0x7fffffffu) : ~u);
}

// ---------------- preprocessing kernels ----------------

__global__ void k_build_edges(const int* __restrict__ ei,
                              int* __restrict__ srcA, int* __restrict__ dstA) {
  int t = blockIdx.x * blockDim.x + threadIdx.x;
  if (t >= EP) return;
  if (t < N_EDGES) { srcA[t] = ei[t]; dstA[t] = ei[N_EDGES + t]; }
  else             { int n = t - N_EDGES; srcA[t] = n; dstA[t] = n; }
}

__global__ void k_loop_accum(const int* __restrict__ dstA, const float* __restrict__ ea,
                             float* __restrict__ deg, float* __restrict__ la) {
  long t = (long)blockIdx.x * blockDim.x + threadIdx.x;
  if (t >= (long)N_EDGES * F_EDGE) return;
  int e = (int)(t >> 5), f = (int)(t & 31);
  int d = dstA[e];
  atomicAdd(&la[(long)d * F_EDGE + f], ea[t]);
  if (f == 0) atomicAdd(&deg[d], 1.0f);
}

__global__ void k_loop_final(const float* __restrict__ deg, const float* __restrict__ la,
                             __bf16* __restrict__ eab) {
  long t = (long)blockIdx.x * blockDim.x + threadIdx.x;
  if (t >= (long)N_NODES * F_EDGE) return;
  int n = (int)(t >> 5);
  float v = la[t] / fmaxf(deg[n], 1.0f);
  eab[(long)(N_EDGES + n) * F_EDGE + (t & 31)] = (__bf16)v;
}

__global__ void k_eab_edges(const float* __restrict__ ea, __bf16* __restrict__ eab) {
  long t = (long)blockIdx.x * blockDim.x + threadIdx.x;
  if (t >= (long)N_EDGES * F_EDGE) return;
  eab[t] = (__bf16)ea[t];
}

__global__ void k_cvt_x(const float* __restrict__ x, __bf16* __restrict__ hb) {
  long t = (long)blockIdx.x * blockDim.x + threadIdx.x;
  if (t >= (long)N_NODES * F_NODE) return;
  hb[t] = (__bf16)x[t];
}

// Pack a K x 128 row-major f32 weight into the per-lane v16bf B layout:
// flat index = ((nt*Kt + kt)*32 + lane)*16 + j
__global__ void k_pack_w(const float* __restrict__ W, __bf16* __restrict__ P, int Kt) {
  int t = blockIdx.x * blockDim.x + threadIdx.x;
  int total = Kt * 8 * 32 * 16;
  if (t >= total) return;
  int j    = t & 15;
  int lane = (t >> 4) & 31;
  int rest = t >> 9;
  int kt   = rest % Kt;
  int nt   = rest / Kt;
  int n = nt * 16 + (lane & 15);
  int k = kt * 32 + ((lane >> 4) * 8) + (j < 8 ? j : j + 8);
  P[t] = (__bf16)W[k * OUTF + n];
}

// ---------------- per-layer kernels ----------------

// xl = hb@Wl + bl ; xr = hb@Wr + br, one wave per 16-row tile, bf16 WMMA, f32 acc.
__global__ void k_gemm_nodes(const __bf16* __restrict__ hb,
                             const __bf16* __restrict__ WlP, const __bf16* __restrict__ WrP,
                             const float* __restrict__ bl, const float* __restrict__ br,
                             float* __restrict__ xl, float* __restrict__ xr) {
  int lane = threadIdx.x & 31;
  int rt = blockIdx.x * (blockDim.x >> 5) + (threadIdx.x >> 5);
  if (rt >= N_NODES / 16) return;
  int hi = lane >> 4;
  int col16 = lane & 15;
  const __bf16* rp = hb + (long)(rt * 16 + col16) * F_NODE;
  bf16x16 A[4];
#pragma unroll
  for (int kt = 0; kt < 4; ++kt)
    A[kt] = load_bf16x16_split(rp, kt * 32 + hi * 8);
#pragma unroll
  for (int which = 0; which < 2; ++which) {
    const __bf16* WP = which ? WrP : WlP;
    const float* bv  = which ? br  : bl;
    float* o         = which ? xr  : xl;
#pragma unroll
    for (int nt = 0; nt < 8; ++nt) {
      float b = bv[nt * 16 + col16];
      f32x8 c;
#pragma unroll
      for (int v = 0; v < 8; ++v) c[v] = b;
#pragma unroll
      for (int kt = 0; kt < 4; ++kt) {
        bf16x16 B = *(const bf16x16*)(WP + ((nt * 4 + kt) * 32 + lane) * 16);
        c = __builtin_amdgcn_wmma_f32_16x16x32_bf16(false, A[kt], false, B,
                                                    (short)0, c, false, false);
      }
      int coln = nt * 16 + col16;
#pragma unroll
      for (int v = 0; v < 8; ++v)
        o[(long)(rt * 16 + v + hi * 8) * OUTF + coln] = c[v];
    }
  }
}

// Fused: eproj = ea@We (WMMA), + xl[src] + xr[dst], LeakyReLU, dot(att) -> score.
// One wave per 16 edges; D-tile lane holds column (lane&15), rows v (+8 for hi half).
__global__ void k_edge_score(const __bf16* __restrict__ eab, const __bf16* __restrict__ WeP,
                             const float* __restrict__ xl, const float* __restrict__ xr,
                             const int* __restrict__ srcA, const int* __restrict__ dstA,
                             const float* __restrict__ att, float* __restrict__ score) {
  int lane = threadIdx.x & 31;
  int et = blockIdx.x * (blockDim.x >> 5) + (threadIdx.x >> 5);
  if (et >= EP / 16) return;
  long base = (long)et * 16;
  int hi = lane >> 4;
  int col16 = lane & 15;
  bf16x16 A = load_bf16x16_split(eab + (base + col16) * F_EDGE, hi * 8);
  int s8[8], d8[8];
#pragma unroll
  for (int v = 0; v < 8; ++v) {
    int m = v + hi * 8;
    s8[v] = srcA[base + m];
    d8[v] = dstA[base + m];
  }
  float acc[8];
#pragma unroll
  for (int v = 0; v < 8; ++v) acc[v] = 0.0f;
#pragma unroll
  for (int nt = 0; nt < 8; ++nt) {
    bf16x16 B = *(const bf16x16*)(WeP + (nt * 32 + lane) * 16);
    f32x8 c;
#pragma unroll
    for (int v = 0; v < 8; ++v) c[v] = 0.0f;
    c = __builtin_amdgcn_wmma_f32_16x16x32_bf16(false, A, false, B,
                                                (short)0, c, false, false);
    int coln = nt * 16 + col16;
    float av = att[coln];
#pragma unroll
    for (int v = 0; v < 8; ++v) {
      float val = c[v] + xl[(long)s8[v] * OUTF + coln] + xr[(long)d8[v] * OUTF + coln];
      val = (val > 0.0f) ? val : SLOPE * val;
      acc[v] += val * av;
    }
  }
#pragma unroll
  for (int v = 0; v < 8; ++v) {
    float a = acc[v];
    a += __shfl_xor(a, 8, 32);
    a += __shfl_xor(a, 4, 32);
    a += __shfl_xor(a, 2, 32);
    a += __shfl_xor(a, 1, 32);
    if (col16 == 0) score[base + v + hi * 8] = a;
  }
}

__global__ void k_segmax(const float* __restrict__ score, const int* __restrict__ dstA,
                         unsigned* __restrict__ segmax) {
  int t = blockIdx.x * blockDim.x + threadIdx.x;
  if (t >= EP) return;
  atomicMax(&segmax[dstA[t]], mono(score[t]));
}

__global__ void k_expsum(float* __restrict__ score, const int* __restrict__ dstA,
                         const unsigned* __restrict__ segmax, float* __restrict__ denom) {
  int t = blockIdx.x * blockDim.x + threadIdx.x;
  if (t >= EP) return;
  int d = dstA[t];
  float pe = __expf(score[t] - demono(segmax[d]));
  score[t] = pe;             // reuse buffer as p
  atomicAdd(&denom[d], pe);
}

__global__ void k_aggregate(const float* __restrict__ p, const float* __restrict__ denom,
                            const int* __restrict__ srcA, const int* __restrict__ dstA,
                            const float* __restrict__ xl, float* __restrict__ hout) {
  long t = (long)blockIdx.x * blockDim.x + threadIdx.x;
  if (t >= (long)EP * OUTF) return;
  int e = (int)(t >> 7), f = (int)(t & 127);
  int d = dstA[e];
  float alpha = p[e] / denom[d];
  atomicAdd(&hout[(long)d * OUTF + f], alpha * xl[(long)srcA[e] * OUTF + f]);
}

__global__ void k_bias_relu(float* __restrict__ hout, const float* __restrict__ bias,
                            __bf16* __restrict__ hb) {
  long t = (long)blockIdx.x * blockDim.x + threadIdx.x;
  if (t >= (long)N_NODES * OUTF) return;
  float v = fmaxf(hout[t] + bias[t & 127], 0.0f);
  hout[t] = v;
  hb[t] = (__bf16)v;
}

// ---------------- readout ----------------

__global__ void k_pool(const float* __restrict__ hout, const int* __restrict__ batch,
                       float* __restrict__ pooled, float* __restrict__ cnt) {
  long t = (long)blockIdx.x * blockDim.x + threadIdx.x;
  if (t >= (long)N_NODES * OUTF) return;
  int n = (int)(t >> 7), f = (int)(t & 127);
  int g = batch[n];
  atomicAdd(&pooled[(long)g * OUTF + f], hout[t]);
  if (f == 0) atomicAdd(&cnt[g], 1.0f);
}

__global__ void k_head(const float* __restrict__ pooled, const float* __restrict__ cnt,
                       const float* __restrict__ lw, const float* __restrict__ lb,
                       float* __restrict__ out) {
  int t = blockIdx.x * blockDim.x + threadIdx.x;
  if (t >= N_GRAPHS * N_CLASSES) return;
  int g = t / N_CLASSES, c = t % N_CLASSES;
  float inv = 1.0f / fmaxf(cnt[g], 1.0f);
  float s = lb[c];
  for (int f = 0; f < OUTF; ++f)
    s += pooled[g * OUTF + f] * inv * lw[f * N_CLASSES + c];
  out[t] = s;
}

// ---------------- host ----------------

static inline unsigned cdivl(long a, long b) { return (unsigned)((a + b - 1) / b); }
static inline char* carve(char*& p, size_t bytes) {
  char* r = p;
  p += (bytes + 255) & ~(size_t)255;
  return r;
}

extern "C" void kernel_launch(void* const* d_in, const int* in_sizes, int n_in,
                              void* d_out, int out_size, void* d_ws, size_t ws_size,
                              hipStream_t stream) {
  (void)in_sizes; (void)n_in; (void)out_size; (void)ws_size;
  const float* x         = (const float*)d_in[0];
  const int*   ei        = (const int*)d_in[1];
  const int*   batch     = (const int*)d_in[2];
  const float* edge_attr = (const float*)d_in[4];
  const float* Wl        = (const float*)d_in[5];
  const float* bl        = (const float*)d_in[6];
  const float* Wr        = (const float*)d_in[7];
  const float* br        = (const float*)d_in[8];
  const float* We        = (const float*)d_in[9];
  const float* att       = (const float*)d_in[10];
  const float* bias      = (const float*)d_in[11];
  const float* lin_w     = (const float*)d_in[12];
  const float* lin_b     = (const float*)d_in[13];
  float* out = (float*)d_out;

  char* p = (char*)d_ws;
  int*      srcA   = (int*)carve(p, (size_t)EP * 4);
  int*      dstA   = (int*)carve(p, (size_t)EP * 4);
  float*    deg    = (float*)carve(p, (size_t)N_NODES * 4);
  float*    la     = (float*)carve(p, (size_t)N_NODES * F_EDGE * 4);
  __bf16*   eab    = (__bf16*)carve(p, (size_t)EP * F_EDGE * 2);
  __bf16*   hb     = (__bf16*)carve(p, (size_t)N_NODES * F_NODE * 2);
  float*    xl     = (float*)carve(p, (size_t)N_NODES * OUTF * 4);
  float*    xr     = (float*)carve(p, (size_t)N_NODES * OUTF * 4);
  float*    hout   = (float*)carve(p, (size_t)N_NODES * OUTF * 4);
  float*    score  = (float*)carve(p, (size_t)EP * 4);
  unsigned* segmax = (unsigned*)carve(p, (size_t)N_NODES * 4);
  float*    denom  = (float*)carve(p, (size_t)N_NODES * 4);
  float*    pooled = (float*)carve(p, (size_t)N_GRAPHS * OUTF * 4);
  float*    cnt    = (float*)carve(p, (size_t)N_GRAPHS * 4);
  __bf16*   WlP    = (__bf16*)carve(p, (size_t)N_LAYERS * 8 * 4 * 32 * 16 * 2);
  __bf16*   WrP    = (__bf16*)carve(p, (size_t)N_LAYERS * 8 * 4 * 32 * 16 * 2);
  __bf16*   WeP    = (__bf16*)carve(p, (size_t)N_LAYERS * 8 * 1 * 32 * 16 * 2);

  const int B = 256;

  // --- preprocessing (self loops, bf16 conversions, weight packing) ---
  hipMemsetAsync(deg, 0, (size_t)N_NODES * 4, stream);
  hipMemsetAsync(la, 0, (size_t)N_NODES * F_EDGE * 4, stream);
  k_build_edges<<<cdivl(EP, B), B, 0, stream>>>(ei, srcA, dstA);
  k_loop_accum<<<cdivl((long)N_EDGES * F_EDGE, B), B, 0, stream>>>(dstA, edge_attr, deg, la);
  k_loop_final<<<cdivl((long)N_NODES * F_EDGE, B), B, 0, stream>>>(deg, la, eab);
  k_eab_edges<<<cdivl((long)N_EDGES * F_EDGE, B), B, 0, stream>>>(edge_attr, eab);
  k_cvt_x<<<cdivl((long)N_NODES * F_NODE, B), B, 0, stream>>>(x, hb);
  for (int l = 0; l < N_LAYERS; ++l) {
    k_pack_w<<<cdivl(4 * 8 * 32 * 16, B), B, 0, stream>>>(Wl + (size_t)l * 128 * 128, WlP + (size_t)l * 16384, 4);
    k_pack_w<<<cdivl(4 * 8 * 32 * 16, B), B, 0, stream>>>(Wr + (size_t)l * 128 * 128, WrP + (size_t)l * 16384, 4);
    k_pack_w<<<cdivl(1 * 8 * 32 * 16, B), B, 0, stream>>>(We + (size_t)l * 32 * 128, WeP + (size_t)l * 4096, 1);
  }

  // --- 3 GATv2 layers ---
  for (int l = 0; l < N_LAYERS; ++l) {
    k_gemm_nodes<<<cdivl(N_NODES / 16, B / 32), B, 0, stream>>>(
        hb, WlP + (size_t)l * 16384, WrP + (size_t)l * 16384,
        bl + l * OUTF, br + l * OUTF, xl, xr);
    hipMemsetAsync(segmax, 0, (size_t)N_NODES * 4, stream);
    hipMemsetAsync(denom, 0, (size_t)N_NODES * 4, stream);
    hipMemsetAsync(hout, 0, (size_t)N_NODES * OUTF * 4, stream);
    k_edge_score<<<cdivl(EP / 16, B / 32), B, 0, stream>>>(
        eab, WeP + (size_t)l * 4096, xl, xr, srcA, dstA, att + l * OUTF, score);
    k_segmax<<<cdivl(EP, B), B, 0, stream>>>(score, dstA, segmax);
    k_expsum<<<cdivl(EP, B), B, 0, stream>>>(score, dstA, segmax, denom);
    k_aggregate<<<cdivl((long)EP * OUTF, B), B, 0, stream>>>(score, denom, srcA, dstA, xl, hout);
    k_bias_relu<<<cdivl((long)N_NODES * OUTF, B), B, 0, stream>>>(hout, bias + l * OUTF, hb);
  }

  // --- readout ---
  hipMemsetAsync(pooled, 0, (size_t)N_GRAPHS * OUTF * 4, stream);
  hipMemsetAsync(cnt, 0, (size_t)N_GRAPHS * 4, stream);
  k_pool<<<cdivl((long)N_NODES * OUTF, B), B, 0, stream>>>(hout, batch, pooled, cnt);
  k_head<<<cdivl(N_GRAPHS * N_CLASSES, B), B, 0, stream>>>(pooled, cnt, lin_w, lin_b, out);
}
